// Bottleneck_5093831213296
// MI455X (gfx1250) — matrix operand order repaired and checked by reading, not compiled, not run on hardware
//
#include <hip/hip_runtime.h>

// CDNA5 / gfx1250 bottleneck block: three convs as implicit GEMMs on
// v_wmma_f32_16x16x32_f16, BN folded to scale/bias, fused ReLU/residual.
// Mainloop: double-buffered LDS tiles filled with GLOBAL_LOAD_ASYNC_TO_LDS_B128
// (ASYNCcnt), one workgroup barrier per 32-wide K-step, 8 WMMAs per wave/step.

typedef __attribute__((ext_vector_type(16))) _Float16 v16h;
typedef __attribute__((ext_vector_type(8)))  _Float16 v8h;
typedef __attribute__((ext_vector_type(8)))  float    v8f;

#define LDS_STRIDE 40   // halves: 80B rows -> bank-conflict-free b128 frag reads
#define MTILE 128
#define NTILE 128
#define KSTEP 32

// ---------------------------------------------------------------- async copy

// Direct global->LDS 16-byte copy, tracked on ASYNCcnt (no VGPR round-trip).
// Low 32 bits of a flat shared pointer == LDS offset (ISA 10.2 aperture rules).
__device__ __forceinline__ void async_b128(const _Float16* g, _Float16* l) {
  unsigned lds = (unsigned)(uintptr_t)l;
  asm volatile("global_load_async_to_lds_b128 %0, %1, off"
               :: "v"(lds), "v"(g) : "memory");
}

__device__ __forceinline__ void wait_async0() {
  asm volatile("s_wait_asynccnt 0x0" ::: "memory");
}

// ---------------------------------------------------------------- fragments

__device__ __forceinline__ v16h make_frag(v8h lo, v8h hi) {
  v16h f;
#pragma unroll
  for (int i = 0; i < 8; ++i) { f[i] = lo[i]; f[i + 8] = hi[i]; }
  return f;
}

// A-matrix 16x32 f16 (ISA 7.12.2): lane L<16 -> row M=L, K {0..7}+{16..23};
// lane L>=16 -> row M=L-16, K {8..15}+{24..31}.
__device__ __forceinline__ v16h load_afrag(const _Float16* As, int mbase, int lane) {
  const _Float16* p = As + (mbase + (lane & 15)) * LDS_STRIDE + ((lane & 16) ? 8 : 0);
  return make_frag(*(const v8h*)p, *(const v8h*)(p + 16));
}

// B-matrix 32x16 f16 from LDS stored [N][K]: lane L -> col N=L&15,
// K {0..7}+{16..23} for L<16, K {8..15}+{24..31} for L>=16 (matches A).
__device__ __forceinline__ v16h load_bfrag(const _Float16* Bs, int nbase, int lane) {
  const _Float16* p = Bs + (nbase + (lane & 15)) * LDS_STRIDE + ((lane & 16) ? 8 : 0);
  return make_frag(*(const v8h*)p, *(const v8h*)(p + 16));
}

// one 128x128x32 step: 2 (M) x 4 (N) wmma per wave
__device__ __forceinline__ void mma_step(const _Float16* As, const _Float16* Bs,
                                         int lane, int wm, int wn, v8f C[2][4]) {
  v16h a0 = load_afrag(As, wm * 32,      lane);
  v16h a1 = load_afrag(As, wm * 32 + 16, lane);
#pragma unroll
  for (int ni = 0; ni < 4; ++ni) {
    v16h b = load_bfrag(Bs, wn * 64 + ni * 16, lane);
    C[0][ni] = __builtin_amdgcn_wmma_f32_16x16x32_f16(false, a0, false, b,
                                                      (short)0, C[0][ni], false, false);
    C[1][ni] = __builtin_amdgcn_wmma_f32_16x16x32_f16(false, a1, false, b,
                                                      (short)0, C[1][ni], false, false);
  }
}

// ---------------------------------------------------------------- prep kernels

__global__ void k_x_to_nhwc_f16(const float* __restrict__ x, _Float16* __restrict__ xh) {
  size_t i = (size_t)blockIdx.x * 256 + threadIdx.x;     // over 25088*1024
  size_t m = i >> 10;
  unsigned c = (unsigned)(i & 1023);
  unsigned n  = (unsigned)(m / 196u);
  unsigned hw = (unsigned)(m - (size_t)n * 196u);
  xh[i] = (_Float16)x[((size_t)n * 1024u + c) * 196u + hw];
}

__global__ void k_f32_to_f16(const float* __restrict__ s, _Float16* __restrict__ d, int n) {
  int i = blockIdx.x * 256 + threadIdx.x;
  if (i < n) d[i] = (_Float16)s[i];
}

// w2 OIHW (256,256,3,3) -> [pos=ky*3+kx][O][C] f16
__global__ void k_pack_w2(const float* __restrict__ w2, _Float16* __restrict__ wp) {
  int i = blockIdx.x * 256 + threadIdx.x;                // over 9*256*256
  int c = i & 255, o = (i >> 8) & 255, pos = i >> 16;
  wp[i] = (_Float16)w2[((size_t)o * 256 + c) * 9 + pos];
}

__global__ void k_bn_prep(const float* __restrict__ g, const float* __restrict__ b,
                          const float* __restrict__ m, const float* __restrict__ v,
                          float* __restrict__ scale, float* __restrict__ bias, int n) {
  int i = blockIdx.x * 256 + threadIdx.x;
  if (i < n) {
    float s = g[i] * rsqrtf(v[i] + 1e-5f);
    scale[i] = s;
    bias[i]  = b[i] - m[i] * s;
  }
}

// ---------------------------------------------------------------- GEMM kernels

// conv1 / conv3 style mainloop: [M x K] @ [Nout x K]^T, K multiple of 32.
// conv1: BN+ReLU -> f16 NHWC out
__global__ __launch_bounds__(256) void k_gemm_bn_relu_f16(
    const _Float16* __restrict__ A, const _Float16* __restrict__ B,
    const float* __restrict__ scale, const float* __restrict__ bias,
    _Float16* __restrict__ out, int K, int Nout) {
  __shared__ __align__(16) _Float16 As[2][MTILE * LDS_STRIDE];
  __shared__ __align__(16) _Float16 Bs[2][NTILE * LDS_STRIDE];
  int tid = threadIdx.x, lane = tid & 31, wid = tid >> 5;
  int wm = wid & 3, wn = wid >> 2;
  int Mb = blockIdx.x * MTILE, Nb = blockIdx.y * NTILE;

  int row = tid >> 1, koff = (tid & 1) << 4;
  const _Float16* ga = A + (size_t)(Mb + row) * K + koff;
  const _Float16* gb = B + (size_t)(Nb + row) * K + koff;
  _Float16* la[2] = { &As[0][row * LDS_STRIDE + koff], &As[1][row * LDS_STRIDE + koff] };
  _Float16* lb[2] = { &Bs[0][row * LDS_STRIDE + koff], &Bs[1][row * LDS_STRIDE + koff] };

  auto fill = [&](int buf, int k0) {
    async_b128(ga + k0,     la[buf]);
    async_b128(ga + k0 + 8, la[buf] + 8);
    async_b128(gb + k0,     lb[buf]);
    async_b128(gb + k0 + 8, lb[buf] + 8);
  };

  v8f C[2][4] = {};
  fill(0, 0);
  wait_async0();
  __syncthreads();
  int nk = K / KSTEP;
  for (int k = 0; k < nk; ++k) {
    if (k + 1 < nk) fill((k + 1) & 1, (k + 1) * KSTEP);
    mma_step(As[k & 1], Bs[k & 1], lane, wm, wn, C);
    wait_async0();
    __syncthreads();
  }

#pragma unroll
  for (int ni = 0; ni < 4; ++ni) {
    int col = Nb + wn * 64 + ni * 16 + (lane & 15);
    float s = scale[col], bb = bias[col];
#pragma unroll
    for (int mi = 0; mi < 2; ++mi) {
      int rbase = Mb + wm * 32 + mi * 16 + ((lane >> 4) << 3);
#pragma unroll
      for (int r = 0; r < 8; ++r) {
        float v = C[mi][ni][r] * s + bb;
        v = v > 0.f ? v : 0.f;
        out[(size_t)(rbase + r) * Nout + col] = (_Float16)v;
      }
    }
  }
}

// conv2: 3x3 pad=1 as 9 shifted GEMMs over NHWC f16, BN+ReLU, f16 out
__global__ __launch_bounds__(256) void k_conv3x3_bn_relu(
    const _Float16* __restrict__ in,   // [25088][256] NHWC f16
    const _Float16* __restrict__ wp,   // [9][256][256] f16
    const float* __restrict__ scale, const float* __restrict__ bias,
    _Float16* __restrict__ out) {
  __shared__ __align__(16) _Float16 As[2][MTILE * LDS_STRIDE];
  __shared__ __align__(16) _Float16 Bs[2][NTILE * LDS_STRIDE];
  int tid = threadIdx.x, lane = tid & 31, wid = tid >> 5;
  int wm = wid & 3, wn = wid >> 2;
  int Mb = blockIdx.x * MTILE, Nb = blockIdx.y * NTILE;

  int row  = tid >> 1;
  int koff = (tid & 1) << 4;
  int m = Mb + row;
  unsigned n  = (unsigned)m / 196u;
  unsigned hw = (unsigned)m - n * 196u;
  int h = (int)(hw / 14u);
  int w = (int)(hw - (unsigned)h * 14u);
  _Float16* la[2] = { &As[0][row * LDS_STRIDE + koff], &As[1][row * LDS_STRIDE + koff] };
  _Float16* lb[2] = { &Bs[0][row * LDS_STRIDE + koff], &Bs[1][row * LDS_STRIDE + koff] };

  auto fill = [&](int buf, int kk) {
    int pos = kk >> 3;
    int c0  = (kk & 7) << 5;
    int ky = pos / 3, kx = pos - ky * 3;
    int hs = h + ky - 1, ws = w + kx - 1;
    const _Float16* gb = wp + (size_t)pos * 65536u + (size_t)(Nb + row) * 256u + c0 + koff;
    async_b128(gb,     lb[buf]);
    async_b128(gb + 8, lb[buf] + 8);
    if (((unsigned)hs < 14u) & ((unsigned)ws < 14u)) {
      const _Float16* ga =
          in + ((size_t)n * 196u + (size_t)(hs * 14 + ws)) * 256u + c0 + koff;
      async_b128(ga,     la[buf]);
      async_b128(ga + 8, la[buf] + 8);
    } else {
      v8h z = {};
      *(v8h*)la[buf]       = z;   // zero padding: plain ds_store (DScnt, barrier-covered)
      *(v8h*)(la[buf] + 8) = z;
    }
  };

  v8f C[2][4] = {};
  fill(0, 0);
  wait_async0();
  __syncthreads();
  for (int kk = 0; kk < 72; ++kk) {          // 9 taps x 8 K-chunks of 32
    if (kk + 1 < 72) fill((kk + 1) & 1, kk + 1);
    mma_step(As[kk & 1], Bs[kk & 1], lane, wm, wn, C);
    wait_async0();
    __syncthreads();
  }

#pragma unroll
  for (int ni = 0; ni < 4; ++ni) {
    int col = Nb + wn * 64 + ni * 16 + (lane & 15);
    float s = scale[col], bb = bias[col];
#pragma unroll
    for (int mi = 0; mi < 2; ++mi) {
      int rbase = Mb + wm * 32 + mi * 16 + ((lane >> 4) << 3);
#pragma unroll
      for (int r = 0; r < 8; ++r) {
        float v = C[mi][ni][r] * s + bb;
        v = v > 0.f ? v : 0.f;
        out[(size_t)(rbase + r) * 256u + col] = (_Float16)v;
      }
    }
  }
}

// conv3: [25088 x 256] @ [1024 x 256]^T, BN + residual(NCHW fp32) + ReLU, fp32 NCHW out
__global__ __launch_bounds__(256) void k_gemm_bn_add_relu_nchw(
    const _Float16* __restrict__ A, const _Float16* __restrict__ B,
    const float* __restrict__ scale, const float* __restrict__ bias,
    const float* __restrict__ xres, float* __restrict__ out) {
  __shared__ __align__(16) _Float16 As[2][MTILE * LDS_STRIDE];
  __shared__ __align__(16) _Float16 Bs[2][NTILE * LDS_STRIDE];
  const int K = 256;
  int tid = threadIdx.x, lane = tid & 31, wid = tid >> 5;
  int wm = wid & 3, wn = wid >> 2;
  int Mb = blockIdx.x * MTILE, Nb = blockIdx.y * NTILE;

  int row = tid >> 1, koff = (tid & 1) << 4;
  const _Float16* ga = A + (size_t)(Mb + row) * K + koff;
  const _Float16* gb = B + (size_t)(Nb + row) * K + koff;
  _Float16* la[2] = { &As[0][row * LDS_STRIDE + koff], &As[1][row * LDS_STRIDE + koff] };
  _Float16* lb[2] = { &Bs[0][row * LDS_STRIDE + koff], &Bs[1][row * LDS_STRIDE + koff] };

  auto fill = [&](int buf, int k0) {
    async_b128(ga + k0,     la[buf]);
    async_b128(ga + k0 + 8, la[buf] + 8);
    async_b128(gb + k0,     lb[buf]);
    async_b128(gb + k0 + 8, lb[buf] + 8);
  };

  v8f C[2][4] = {};
  fill(0, 0);
  wait_async0();
  __syncthreads();
  const int nk = K / KSTEP;
  for (int k = 0; k < nk; ++k) {
    if (k + 1 < nk) fill((k + 1) & 1, (k + 1) * KSTEP);
    mma_step(As[k & 1], Bs[k & 1], lane, wm, wn, C);
    wait_async0();
    __syncthreads();
  }

#pragma unroll
  for (int ni = 0; ni < 4; ++ni) {
    int col = Nb + wn * 64 + ni * 16 + (lane & 15);
    float s = scale[col], bb = bias[col];
#pragma unroll
    for (int mi = 0; mi < 2; ++mi) {
      int rbase = Mb + wm * 32 + mi * 16 + ((lane >> 4) << 3);
#pragma unroll
      for (int r = 0; r < 8; ++r) {
        unsigned mm = (unsigned)(rbase + r);
        unsigned nn = mm / 196u;
        unsigned hw = mm - nn * 196u;
        size_t idx = ((size_t)nn * 1024u + (unsigned)col) * 196u + hw;
        float v = C[mi][ni][r] * s + bb + xres[idx];
        out[idx] = v > 0.f ? v : 0.f;
      }
    }
  }
}

// ---------------------------------------------------------------- launch

extern "C" void kernel_launch(void* const* d_in, const int* in_sizes, int n_in,
                              void* d_out, int out_size, void* d_ws, size_t ws_size,
                              hipStream_t stream) {
  const float* x  = (const float*)d_in[0];
  const float* w1 = (const float*)d_in[1];
  const float* w2 = (const float*)d_in[2];
  const float* w3 = (const float*)d_in[3];
  const float* g1 = (const float*)d_in[4];  const float* b1 = (const float*)d_in[5];
  const float* m1 = (const float*)d_in[6];  const float* v1 = (const float*)d_in[7];
  const float* g2 = (const float*)d_in[8];  const float* b2 = (const float*)d_in[9];
  const float* m2 = (const float*)d_in[10]; const float* v2 = (const float*)d_in[11];
  const float* g3 = (const float*)d_in[12]; const float* b3 = (const float*)d_in[13];
  const float* m3 = (const float*)d_in[14]; const float* v3 = (const float*)d_in[15];

  char* ws = (char*)d_ws;
  _Float16* xh  = (_Float16*)(ws + 0);           // 25088*1024 f16 = 51,380,224 B
  _Float16* o1  = (_Float16*)(ws + 51380224);    // 25088*256  f16 = 12,845,056 B
  _Float16* o2  = (_Float16*)(ws + 64225280);    // 25088*256  f16
  _Float16* w1h = (_Float16*)(ws + 77070336);    // 256*1024   f16 = 524,288 B
  _Float16* w2p = (_Float16*)(ws + 77594624);    // 9*256*256  f16 = 1,179,648 B
  _Float16* w3h = (_Float16*)(ws + 78774272);    // 1024*256   f16 = 524,288 B
  float* s1  = (float*)(ws + 79298560);
  float* bb1 = s1 + 256;
  float* s2  = bb1 + 256;
  float* bb2 = s2 + 256;
  float* s3  = bb2 + 256;
  float* bb3 = s3 + 1024;
  float* outp = (float*)d_out;

  k_x_to_nhwc_f16<<<100352, 256, 0, stream>>>(x, xh);
  k_f32_to_f16<<<1024, 256, 0, stream>>>(w1, w1h, 262144);
  k_f32_to_f16<<<1024, 256, 0, stream>>>(w3, w3h, 262144);
  k_pack_w2<<<2304, 256, 0, stream>>>(w2, w2p);
  k_bn_prep<<<1, 256, 0, stream>>>(g1, b1, m1, v1, s1, bb1, 256);
  k_bn_prep<<<1, 256, 0, stream>>>(g2, b2, m2, v2, s2, bb2, 256);
  k_bn_prep<<<4, 256, 0, stream>>>(g3, b3, m3, v3, s3, bb3, 1024);

  k_gemm_bn_relu_f16<<<dim3(196, 2), 256, 0, stream>>>(xh, w1h, s1, bb1, o1, 1024, 256);
  k_conv3x3_bn_relu<<<dim3(196, 2), 256, 0, stream>>>(o1, w2p, s2, bb2, o2);
  k_gemm_bn_add_relu_nchw<<<dim3(196, 8), 256, 0, stream>>>(o2, w3h, s3, bb3, x, outp);
}
